// PaperLSTMLayer_67800353735261
// MI455X (gfx1250) — matrix operand order repaired and checked by reading, not compiled, not run on hardware
//
#include <hip/hip_runtime.h>
#include <hip/hip_bf16.h>
#include <stdint.h>

// LSTM: B=4096, D=2048, H=2048, C=4096. gates = concat(x,h) @ [Wi;Wf;Wg;Wo]^T + b
// Pass1: bf16 pack of A=[B,C]. Pass2: bf16 transpose W -> Wt=[C,4H].
// Pass3: WMMA bf16 GEMM (wave tile M=32, N=16 x 4 gates), double-buffered LDS
//        B tiles filled with global_load_async_to_lds_b128, fused LSTM epilogue.

typedef __attribute__((ext_vector_type(16))) __bf16 v16bf;
typedef __attribute__((ext_vector_type(8)))  __bf16 v8bf;
typedef __attribute__((ext_vector_type(8)))  float  v8f;
typedef __attribute__((address_space(3)))    uint16_t lds_u16;

#define B_SZ   4096
#define D_SZ   2048
#define H_SZ   2048
#define C_SZ   (D_SZ + H_SZ)     // 4096
#define NGATE  4
#define WT_COLS (NGATE * H_SZ)   // 8192
#define BTILE  (NGATE * 32 * 16) // bf16 elements per B tile (4 KB)

__device__ __forceinline__ uint16_t f2bf(float f) {
    uint32_t u = __float_as_uint(f);
    uint32_t r = (u + 0x7FFFu + ((u >> 16) & 1u)) >> 16;  // RNE
    return (uint16_t)r;
}

// branch-free gate activations on v_exp_f32 (base-2) + v_rcp_f32
__device__ __forceinline__ float fast_sigmoid(float x) {
    return __builtin_amdgcn_rcpf(1.0f + __builtin_amdgcn_exp2f(-1.4426950408889634f * x));
}
__device__ __forceinline__ float fast_tanh(float x) {
    return 1.0f - 2.0f * __builtin_amdgcn_rcpf(1.0f + __builtin_amdgcn_exp2f(2.8853900817779268f * x));
}

// ---------------------------------------------------------------------------
__global__ void pack_concat_bf16(const float* __restrict__ x,
                                 const float* __restrict__ h,
                                 uint16_t* __restrict__ A) {
    size_t idx = (size_t)blockIdx.x * blockDim.x + threadIdx.x;   // < B*C
    int b = (int)(idx / C_SZ);
    int k = (int)(idx % C_SZ);
    float v = (k < D_SZ) ? x[(size_t)b * D_SZ + k]
                         : h[(size_t)b * H_SZ + (k - D_SZ)];
    A[idx] = f2bf(v);
}

// ---------------------------------------------------------------------------
__global__ void transpose_w_bf16(const float* __restrict__ Wi,
                                 const float* __restrict__ Wf,
                                 const float* __restrict__ Wg,
                                 const float* __restrict__ Wo,
                                 uint16_t* __restrict__ Wt) {
    __shared__ float tile[32][33];
    const int kbase = blockIdx.x * 32;
    const int gate  = blockIdx.y >> 6;            // H/32 == 64 tiles per gate
    const int hbase = (blockIdx.y & 63) * 32;
    const float* W = (gate == 0) ? Wi : (gate == 1) ? Wf : (gate == 2) ? Wg : Wo;
    const int tx = threadIdx.x, ty = threadIdx.y;
#pragma unroll
    for (int i = 0; i < 4; ++i) {
        int hh = hbase + ty + i * 8;
        tile[ty + i * 8][tx] = W[(size_t)hh * C_SZ + kbase + tx];
    }
    __syncthreads();
#pragma unroll
    for (int i = 0; i < 4; ++i) {
        int kk = kbase + ty + i * 8;
        Wt[(size_t)kk * WT_COLS + (size_t)gate * H_SZ + hbase + tx] =
            f2bf(tile[tx][ty + i * 8]);
    }
}

// ---------------------------------------------------------------------------
__device__ __forceinline__ v16bf load_afrag(const uint16_t* __restrict__ arow,
                                            int kbase, int halfid) {
    v8bf a0 = *(const v8bf*)&arow[kbase + halfid * 8];
    v8bf a1 = *(const v8bf*)&arow[kbase + 16 + halfid * 8];
    v16bf r;
#pragma unroll
    for (int i = 0; i < 8; ++i) { r[i] = a0[i]; r[i + 8] = a1[i]; }
    return r;
}

__global__ __launch_bounds__(256)
void lstm_gemm_wmma(const uint16_t* __restrict__ A,
                    const uint16_t* __restrict__ Wt,
                    const float* __restrict__ bi,
                    const float* __restrict__ bfv,
                    const float* __restrict__ bg,
                    const float* __restrict__ bo,
                    const float* __restrict__ cprev,
                    float* __restrict__ out) {
    __shared__ __align__(16) uint16_t ldsB[2 * BTILE];   // 8 KB, double buffered

    const int tid  = threadIdx.x;
    const int wave = tid >> 5;
    const int lane = tid & 31;
    const int m0   = blockIdx.x * 256 + wave * 32;       // 32 rows per wave
    const int n0   = blockIdx.y * 16;

    const v8f vzero = {0.f, 0.f, 0.f, 0.f, 0.f, 0.f, 0.f, 0.f};
    v8f acc[2][NGATE];
#pragma unroll
    for (int f = 0; f < 2; ++f)
#pragma unroll
        for (int g = 0; g < NGATE; ++g) acc[f][g] = vzero;

    const int halfid = lane >> 4;
    const uint16_t* arow0 = A + (size_t)(m0 + (lane & 15)) * C_SZ;
    const uint16_t* arow1 = arow0 + (size_t)16 * C_SZ;

    // cooperative async B staging: 256 threads x 16B = one 4 KB tile
    const int gq   = tid >> 6;       // gate 0..3
    const int rem  = tid & 63;
    const int kk   = rem >> 1;       // 0..31
    const int half = rem & 1;
    const uint32_t lds_elt = (uint32_t)((gq * 32 + kk) * 16 + half * 8);

    // prologue: stage tile for kbase=0 into buffer 0
    {
        uint32_t voff = ((uint32_t)kk * WT_COLS + (uint32_t)gq * H_SZ +
                         (uint32_t)(n0 + half * 8)) * 2u;
        uint32_t loff = (uint32_t)(uintptr_t)(lds_u16*)&ldsB[lds_elt];
        asm volatile("global_load_async_to_lds_b128 %0, %1, %2"
                     :: "v"(loff), "v"(voff), "s"(Wt) : "memory");
        asm volatile("s_wait_asynccnt 0x0" ::: "memory");
        __syncthreads();
    }

    const int NITER = C_SZ / 32;     // 128
#pragma unroll 1
    for (int it = 0; it < NITER; ++it) {
        const int kbase = it * 32;
        const uint32_t rd = (uint32_t)(it & 1) * BTILE;
        const uint32_t wr = BTILE - rd;

        // stage next K-tile into the other buffer (async, overlaps WMMAs)
        if (it + 1 < NITER) {
            uint32_t voff = ((uint32_t)(kbase + 32 + kk) * WT_COLS +
                             (uint32_t)gq * H_SZ + (uint32_t)(n0 + half * 8)) * 2u;
            uint32_t loff = (uint32_t)(uintptr_t)(lds_u16*)&ldsB[wr + lds_elt];
            asm volatile("global_load_async_to_lds_b128 %0, %1, %2"
                         :: "v"(loff), "v"(voff), "s"(Wt) : "memory");
        }

        // A fragments (ISA 16-bit 16x32 layout), 2 M-tiles
        v16bf a0 = load_afrag(arow0, kbase, halfid);
        v16bf a1 = load_afrag(arow1, kbase, halfid);

        // B fragments: lane L holds N=L%16; element s holds K in table order (same as A)
        v16bf bfrag[NGATE];
#pragma unroll
        for (int g = 0; g < NGATE; ++g)
#pragma unroll
            for (int s = 0; s < 16; ++s)
                bfrag[g][s] = *(const __bf16*)&ldsB[rd + (uint32_t)(g * 32 + (s < 8 ? halfid * 8 + s : 16 + halfid * 8 + (s - 8))) * 16 + (lane & 15)];

#pragma unroll
        for (int g = 0; g < NGATE; ++g)
            acc[0][g] = __builtin_amdgcn_wmma_f32_16x16x32_bf16(
                false, a0, false, bfrag[g], (short)0, acc[0][g], false, false);
#pragma unroll
        for (int g = 0; g < NGATE; ++g)
            acc[1][g] = __builtin_amdgcn_wmma_f32_16x16x32_bf16(
                false, a1, false, bfrag[g], (short)0, acc[1][g], false, false);

        asm volatile("s_wait_asynccnt 0x0" ::: "memory");
        __syncthreads();
    }

    // Fused LSTM epilogue. C/D layout: VGPR v, lanes 0-15 -> row v, 16-31 -> v+8
    const int col = n0 + (lane & 15);
    const float vbi = bi[col], vbf = bfv[col], vbg = bg[col], vbo = bo[col];
#pragma unroll
    for (int f = 0; f < 2; ++f) {
        const int rowbase = m0 + f * 16 + 8 * halfid;
#pragma unroll
        for (int v = 0; v < 8; ++v) {
            const int row = rowbase + v;
            float it = fast_sigmoid(acc[f][0][v] + vbi);
            float ft = fast_sigmoid(acc[f][1][v] + vbf);
            float gt = fast_tanh(acc[f][2][v] + vbg);
            float ot = fast_sigmoid(acc[f][3][v] + vbo);
            float cp = cprev[(size_t)row * H_SZ + col];
            float ct = ft * cp + it * gt;
            float ht = ot * fast_tanh(ct);
            out[(size_t)row * H_SZ + col] = ht;                          // h_t
            out[(size_t)B_SZ * H_SZ + (size_t)row * H_SZ + col] = ct;    // c_t
        }
    }
}

// ---------------------------------------------------------------------------
extern "C" void kernel_launch(void* const* d_in, const int* in_sizes, int n_in,
                              void* d_out, int out_size, void* d_ws, size_t ws_size,
                              hipStream_t stream) {
    const float* x  = (const float*)d_in[0];
    const float* hp = (const float*)d_in[1];
    const float* cp = (const float*)d_in[2];
    const float* Wi = (const float*)d_in[3];
    const float* bi = (const float*)d_in[4];
    const float* Wf = (const float*)d_in[5];
    const float* bf = (const float*)d_in[6];
    const float* Wg = (const float*)d_in[7];
    const float* bg = (const float*)d_in[8];
    const float* Wo = (const float*)d_in[9];
    const float* bo = (const float*)d_in[10];
    float* out = (float*)d_out;

    uint16_t* Abf = (uint16_t*)d_ws;                                    // 32 MB
    uint16_t* Wt  = (uint16_t*)((char*)d_ws + (size_t)B_SZ * C_SZ * 2); // 64 MB

    pack_concat_bf16<<<(B_SZ * C_SZ) / 256, 256, 0, stream>>>(x, hp, Abf);
    transpose_w_bf16<<<dim3(C_SZ / 32, WT_COLS / 32), dim3(32, 8), 0, stream>>>(
        Wi, Wf, Wg, Wo, Wt);
    lstm_gemm_wmma<<<dim3(B_SZ / 256, H_SZ / 16), 256, 0, stream>>>(
        Abf, Wt, bi, bf, bg, bo, cp, out);
}
